// MultiHeadAttention_16509854285979
// MI455X (gfx1250) — matrix operand order repaired and checked by reading, not compiled, not run on hardware
//
#include <hip/hip_runtime.h>
#include <hip/hip_bf16.h>

// MHA forward for B=16, N=1024, D=1024, H=16, dk=64 on gfx1250 (wave32, WMMA).
// Pipeline: cvt(x)->bf16, transpose(W)->bf16, QKV GEMM (wmma bf16),
// flash-attention (wmma bf16 + online softmax), FC GEMM (wmma bf16, f32 out).

typedef __attribute__((ext_vector_type(16))) __bf16 v16bf;
typedef __attribute__((ext_vector_type(8)))  __bf16 v8bf;
typedef __attribute__((ext_vector_type(8)))  float  v8f;

__device__ __forceinline__ v16bf cat8(v8bf lo, v8bf hi) {
  return __builtin_shufflevector(lo, hi, 0,1,2,3,4,5,6,7,8,9,10,11,12,13,14,15);
}

// A-fragment (16x32, bf16), row-major source with leading dim lda.
// Lane L holds row (rowbase + L%16); elements 0..7 = K[k0+kb .. +7],
// elements 8..15 = K[k0+kb+16 .. +23], kb = 8*(L/16).  (ISA 7.12.2)
__device__ __forceinline__ v16bf load_a_frag(const __bf16* __restrict__ base,
                                             int lda, int rowbase, int k0, int lane) {
  const int r  = rowbase + (lane & 15);
  const int kb = (lane >> 4) * 8;
  const __bf16* p = base + (size_t)r * lda + k0 + kb;
  v8bf lo = *(const v8bf*)(p);
  v8bf hi = *(const v8bf*)(p + 16);
  return cat8(lo, hi);
}

// B-fragment (32x16, bf16) from a TRANSPOSED operand BT[n][k] (ldb = K stride).
// Lane L holds column n = (colbase + L%16); elements 0..7 = K[k0+kb .. +7],
// elements 8..15 = K[k0+kb+16 .. +23], kb = 8*(L/16) (matches A-fragment).
__device__ __forceinline__ v16bf load_bt_frag(const __bf16* __restrict__ baseT,
                                              int ldb, int colbase, int k0, int lane) {
  const int n    = colbase + (lane & 15);
  const int koff = (lane >> 4) * 8;
  const __bf16* p = baseT + (size_t)n * ldb + k0 + koff;
  return cat8(*(const v8bf*)(p), *(const v8bf*)(p + 16));
}

__device__ __forceinline__ v8f wmma_bf16(v16bf a, v16bf b, v8f c) {
  return __builtin_amdgcn_wmma_f32_16x16x32_bf16(false, a, false, b, (short)0, c, false, false);
}

// ---------------------------------------------------------------- conversions
__global__ void __launch_bounds__(256)
cvt_f32_to_bf16(const float* __restrict__ in, __bf16* __restrict__ out, int n) {
  int i = blockIdx.x * 256 + threadIdx.x;
  if (i < n) out[i] = (__bf16)in[i];
}

// in: [rows][cols] f32  ->  out: [cols][rows] bf16  (transposed)
__global__ void __launch_bounds__(256)
transpose_to_bf16(const float* __restrict__ in, __bf16* __restrict__ out,
                  int rows, int cols) {
  int i = blockIdx.x * 256 + threadIdx.x;
  if (i >= rows * cols) return;
  int r = i / cols, c = i - r * cols;
  out[(size_t)c * rows + r] = (__bf16)in[i];
}

// ---------------------------------------------------------------- QKV GEMM
// C[16384,3072] = xb[16384,1024] @ WqkvT[3072,1024]^T + b ; scatter to Q/K/Vt.
__global__ void __launch_bounds__(256)
gemm_qkv_kernel(const __bf16* __restrict__ A, const __bf16* __restrict__ Wt,
                const float* __restrict__ bias,
                __bf16* __restrict__ Qb, __bf16* __restrict__ Kb,
                __bf16* __restrict__ Vt) {
  const int lane = threadIdx.x & 31;
  const int wave = threadIdx.x >> 5;
  const int tile = blockIdx.x * 8 + wave;          // 512*48 = 24576 tiles
  if (tile >= 512 * 48) return;
  const int tm = tile % 512, tn = tile / 512;
  const int mbase = tm * 32, nbase = tn * 64;

  v8f acc[2][4] = {};
  for (int k0 = 0; k0 < 1024; k0 += 32) {
    if (k0 + 32 < 1024)
      __builtin_prefetch(A + (size_t)(mbase + (lane & 15)) * 1024 + k0 + 32, 0, 1);
    v16bf a0 = load_a_frag(A, 1024, mbase,      k0, lane);
    v16bf a1 = load_a_frag(A, 1024, mbase + 16, k0, lane);
#pragma unroll
    for (int ni = 0; ni < 4; ++ni) {
      v16bf b = load_bt_frag(Wt, 1024, nbase + 16 * ni, k0, lane);
      acc[0][ni] = wmma_bf16(a0, b, acc[0][ni]);
      acc[1][ni] = wmma_bf16(a1, b, acc[1][ni]);
    }
  }
  // Epilogue: n-tile (64 wide) lies entirely inside one of {Q,K,V} and one head.
  const int which = nbase >> 10;                   // 0=Q 1=K 2=V
  const int h     = (nbase & 1023) >> 6;
#pragma unroll
  for (int ni = 0; ni < 4; ++ni) {
    const int dim = 16 * ni + (lane & 15);
    const float bv = bias[nbase + dim];
#pragma unroll
    for (int mi = 0; mi < 2; ++mi) {
#pragma unroll
      for (int r = 0; r < 8; ++r) {
        const int m = mbase + 16 * mi + r + 8 * (lane >> 4);
        const int bidx = m >> 10, row = m & 1023;
        const __bf16 v = (__bf16)(acc[mi][ni][r] + bv);
        const size_t bh = (size_t)(bidx * 16 + h);
        if (which == 0)      Qb[(bh * 1024 + row) * 64 + dim] = v;
        else if (which == 1) Kb[(bh * 1024 + row) * 64 + dim] = v;
        else                 Vt[(bh * 64 + dim) * 1024 + row] = v;
      }
    }
  }
}

// ---------------------------------------------------------------- attention
// One wave per 16-query tile; flash-style online softmax over 32-key tiles.
__global__ void __launch_bounds__(256)
attn_kernel(const __bf16* __restrict__ Qb, const __bf16* __restrict__ Kb,
            const __bf16* __restrict__ Vt, __bf16* __restrict__ ctx) {
  __shared__ __align__(16) __bf16 lds_p[8][16 * 32];
  const int lane = threadIdx.x & 31;
  const int wave = threadIdx.x >> 5;
  __bf16* P = lds_p[wave];
  const int task = blockIdx.x * 8 + wave;          // 256 bh * 64 qtiles = 16384
  if (task >= 16384) return;
  const int qt = task & 63;
  const int bh = task >> 6;
  const __bf16* Q = Qb + (size_t)bh * 1024 * 64;
  const __bf16* K = Kb + (size_t)bh * 1024 * 64;
  const __bf16* V = Vt + (size_t)bh * 64 * 1024;   // [64][1024]
  const int qbase = qt * 16;

  const v16bf qa0 = load_a_frag(Q, 64, qbase, 0,  lane);
  const v16bf qa1 = load_a_frag(Q, 64, qbase, 32, lane);

  float rmax[8], rsum[8];
#pragma unroll
  for (int r = 0; r < 8; ++r) { rmax[r] = -1e30f; rsum[r] = 0.f; }
  v8f acc[4] = {};
  const float scale = 0.125f;                      // 1/sqrt(64)

  for (int kt = 0; kt < 1024; kt += 32) {
    // S(16x32) = Q(16x64) @ K^T : 4 WMMAs
    v8f s0 = {}, s1 = {};
    {
      v16bf b00 = load_bt_frag(K, 64, kt,      0,  lane);
      v16bf b01 = load_bt_frag(K, 64, kt,      32, lane);
      s0 = wmma_bf16(qa0, b00, s0);
      s0 = wmma_bf16(qa1, b01, s0);
      v16bf b10 = load_bt_frag(K, 64, kt + 16, 0,  lane);
      v16bf b11 = load_bt_frag(K, 64, kt + 16, 32, lane);
      s1 = wmma_bf16(qa0, b10, s1);
      s1 = wmma_bf16(qa1, b11, s1);
    }
    // Online softmax. C-frag row r lives on a fixed 16-lane half-group, so
    // xor-butterflies with masks 1,2,4,8 reduce exactly over one S row.
    float p0[8], p1[8], resc[8];
#pragma unroll
    for (int r = 0; r < 8; ++r) {
      float a = s0[r] * scale;
      float b = s1[r] * scale;
      float m = fmaxf(a, b);
#pragma unroll
      for (int off = 1; off < 16; off <<= 1)
        m = fmaxf(m, __shfl_xor(m, off, 32));
      const float nm = fmaxf(rmax[r], m);
      resc[r] = __expf(rmax[r] - nm);
      p0[r] = __expf(a - nm);
      p1[r] = __expf(b - nm);
      float ps = p0[r] + p1[r];
#pragma unroll
      for (int off = 1; off < 16; off <<= 1)
        ps += __shfl_xor(ps, off, 32);
      rsum[r] = rsum[r] * resc[r] + ps;
      rmax[r] = nm;
    }
#pragma unroll
    for (int ni = 0; ni < 4; ++ni)
#pragma unroll
      for (int r = 0; r < 8; ++r) acc[ni][r] *= resc[r];

    // Stage P (C-layout) into LDS row-major 16x32, reload as A-fragment.
#pragma unroll
    for (int r = 0; r < 8; ++r) {
      const int row = r + 8 * (lane >> 4);
      P[row * 32 + (lane & 15)]      = (__bf16)p0[r];
      P[row * 32 + 16 + (lane & 15)] = (__bf16)p1[r];
    }
    asm volatile("" ::: "memory");   // DS ops are in-order within a wave
    v16bf pa;
    {
      const int row = lane & 15;
      const int kb  = 8 * (lane >> 4);
      v8bf lo = *(const v8bf*)(P + row * 32 + kb);
      v8bf hi = *(const v8bf*)(P + row * 32 + kb + 16);
      pa = cat8(lo, hi);
    }
    // context(16x64) += P(16x32) @ V(32x64) : 4 WMMAs
#pragma unroll
    for (int ni = 0; ni < 4; ++ni) {
      v16bf vb = load_bt_frag(V, 1024, 16 * ni, kt, lane);
      acc[ni] = wmma_bf16(pa, vb, acc[ni]);
    }
    asm volatile("" ::: "memory");   // keep next tile's P stores after reads
  }

  // Finalize: divide by softmax denominator, write ctx[B*N, D] bf16.
  const int b = bh >> 4, h = bh & 15;
#pragma unroll
  for (int ni = 0; ni < 4; ++ni) {
    const int dim = 16 * ni + (lane & 15);
#pragma unroll
    for (int r = 0; r < 8; ++r) {
      const int qrow = qbase + r + 8 * (lane >> 4);
      const float v = acc[ni][r] / rsum[r];
      ctx[((size_t)b * 1024 + qrow) * 1024 + h * 64 + dim] = (__bf16)v;
    }
  }
}

// ---------------------------------------------------------------- FC GEMM
// out[16384,1024] = ctx[16384,1024] @ WfcT[1024,1024]^T + b_fc   (f32 out)
__global__ void __launch_bounds__(256)
gemm_fc_kernel(const __bf16* __restrict__ A, const __bf16* __restrict__ Wt,
               const float* __restrict__ bias, float* __restrict__ out) {
  const int lane = threadIdx.x & 31;
  const int wave = threadIdx.x >> 5;
  const int tile = blockIdx.x * 8 + wave;          // 512*16 = 8192 tiles
  if (tile >= 512 * 16) return;
  const int tm = tile % 512, tn = tile / 512;
  const int mbase = tm * 32, nbase = tn * 64;

  v8f acc[2][4] = {};
  for (int k0 = 0; k0 < 1024; k0 += 32) {
    if (k0 + 32 < 1024)
      __builtin_prefetch(A + (size_t)(mbase + (lane & 15)) * 1024 + k0 + 32, 0, 1);
    v16bf a0 = load_a_frag(A, 1024, mbase,      k0, lane);
    v16bf a1 = load_a_frag(A, 1024, mbase + 16, k0, lane);
#pragma unroll
    for (int ni = 0; ni < 4; ++ni) {
      v16bf b = load_bt_frag(Wt, 1024, nbase + 16 * ni, k0, lane);
      acc[0][ni] = wmma_bf16(a0, b, acc[0][ni]);
      acc[1][ni] = wmma_bf16(a1, b, acc[1][ni]);
    }
  }
#pragma unroll
  for (int ni = 0; ni < 4; ++ni) {
    const int n = nbase + 16 * ni + (lane & 15);
    const float bv = bias[n];
#pragma unroll
    for (int mi = 0; mi < 2; ++mi) {
#pragma unroll
      for (int r = 0; r < 8; ++r) {
        const int m = mbase + 16 * mi + r + 8 * (lane >> 4);
        out[(size_t)m * 1024 + n] = acc[mi][ni][r] + bv;
      }
    }
  }
}

// ---------------------------------------------------------------- launch
extern "C" void kernel_launch(void* const* d_in, const int* in_sizes, int n_in,
                              void* d_out, int out_size, void* d_ws, size_t ws_size,
                              hipStream_t stream) {
  const float* x     = (const float*)d_in[0];   // [16,1024,1024]
  const float* W_qkv = (const float*)d_in[1];   // [1024,3072]
  const float* b_qkv = (const float*)d_in[2];   // [3072]
  const float* W_fc  = (const float*)d_in[3];   // [1024,1024]
  const float* b_fc  = (const float*)d_in[4];   // [1024]
  float* out = (float*)d_out;                   // [16,1024,1024]

  char* ws = (char*)d_ws;
  __bf16* xb  = (__bf16*)(ws);                  // 32 MB  [16384][1024]
  __bf16* ctx = xb;                             // alias: xb dead after QKV GEMM
  __bf16* WqT = (__bf16*)(ws + 33554432);       //  6 MB  [3072][1024]
  __bf16* WfT = (__bf16*)(ws + 39845888);       //  2 MB  [1024][1024]
  __bf16* Qb  = (__bf16*)(ws + 41943040);       // 32 MB  [B,H,1024,64]
  __bf16* Kb  = (__bf16*)(ws + 75497472);       // 32 MB  [B,H,1024,64]
  __bf16* Vt  = (__bf16*)(ws + 109051904);      // 32 MB  [B,H,64,1024]

  cvt_f32_to_bf16<<<16777216 / 256, 256, 0, stream>>>(x, xb, 16777216);
  transpose_to_bf16<<<(1024 * 3072) / 256, 256, 0, stream>>>(W_qkv, WqT, 1024, 3072);
  transpose_to_bf16<<<(1024 * 1024) / 256, 256, 0, stream>>>(W_fc, WfT, 1024, 1024);
  gemm_qkv_kernel<<<3072, 256, 0, stream>>>(xb, WqT, b_qkv, Qb, Kb, Vt);
  attn_kernel<<<2048, 256, 0, stream>>>(Qb, Kb, Vt, ctx);
  gemm_fc_kernel<<<1024, 256, 0, stream>>>(ctx, WfT, b_fc, out);
}
